// SwinBlockV2_56401510531724
// MI455X (gfx1250) — matrix operand order repaired and checked by reading, not compiled, not run on hardware
//
#include <hip/hip_runtime.h>
#include <math.h>

// ---------------------------------------------------------------------------
// Swin-V2 block for MI455X (gfx1250, wave32, WMMA bf16 -> f32 accumulate,
// async global->LDS double-buffered GEMM)
// ---------------------------------------------------------------------------

typedef __attribute__((ext_vector_type(16))) __bf16        v16bf;
typedef __attribute__((ext_vector_type(8)))  float         v8f;
typedef __attribute__((ext_vector_type(4)))  unsigned int  u32x4;
typedef unsigned short u16;

union FragAB { u32x4 u[2]; v16bf v; };   // 32B: one WMMA A/B operand
union Pack8  { u32x4 u; u16 s[8]; };

__device__ __forceinline__ u16 f2bf(float f) {
  unsigned int u = __float_as_uint(f);
  u += 0x7FFFu + ((u >> 16) & 1u);          // round-to-nearest-even
  return (u16)(u >> 16);
}
__device__ __forceinline__ float bf2f(u16 h) {
  return __uint_as_float(((unsigned int)h) << 16);
}

// CDNA5 async global->LDS (ASYNCcnt path, see CDNA5 ISA).
// Low 32 bits of a flat __shared__ pointer == LDS byte offset (ISA 10.2).
__device__ __forceinline__ void async_ld_b128(u16* lds_dst, const u16* gsrc) {
  unsigned lds = (unsigned)(size_t)lds_dst;
  asm volatile("global_load_async_to_lds_b128 %0, %1, off"
               :: "v"(lds), "v"(gsrc)
               : "memory");
}
__device__ __forceinline__ void wait_async0() {
  asm volatile("s_wait_asynccnt 0x0" ::: "memory");
}
__device__ __forceinline__ void wait_async4() {
  asm volatile("s_wait_asynccnt 0x4" ::: "memory");
}

// problem dims
#define DIMC   384
#define NHEAD  12
#define HDIM   32
#define NTOK   49      // 7x7 tokens per window
#define NPAD   64      // padded window rows
#define TOTWIN 2048    // 32 batches * 64 windows
#define TOKENS 100352  // 32 * 3136
#define MROWS  131072  // TOTWIN * NPAD

// workspace layout (bytes)
static const size_t OFF_WQKVT  = 0;          // (1152x384) bf16 transposed  884736
static const size_t OFF_WPROJT = 884736;     // (384x384)  bf16 transposed  294912
static const size_t OFF_W1T    = 1179648;    // (1536x384) bf16 transposed 1179648
static const size_t OFF_W2T    = 2359296;    // (384x1536) bf16 transposed 1179648
static const size_t OFF_BIAS   = 3538944;    // crpb bias (12,49,49) f32    115248
static const size_t OFF_XW     = 3654400;    // Xw bf16 (2048,64,384); later attn_out
static const size_t OFF_QKV    = 104317696;  // qkv bf16; later Y f32 / mlp-hidden bf16
static const size_t OFF_X1     = 412599040;  // x1 f32 (100352,384)
static const size_t OFF_LN2    = 566739712;  // ln2 bf16 (100352,384)

// ---------------------------------------------------------------------------
// misc small kernels
// ---------------------------------------------------------------------------
__global__ void k_zero16(u32x4* __restrict__ p, long n16) {
  long i = (long)blockIdx.x * blockDim.x + threadIdx.x;
  long stride = (long)gridDim.x * blockDim.x;
  u32x4 z = {0u, 0u, 0u, 0u};
  for (; i < n16; i += stride) p[i] = z;
}

// W (KxN) f32 row-major -> Wt (NxK) bf16 row-major
__global__ void k_wtrans(const float* __restrict__ W, u16* __restrict__ Wt,
                         int K, int N) {
  long i = (long)blockIdx.x * 256 + threadIdx.x;
  if (i >= (long)K * N) return;
  int k = (int)(i / N), n = (int)(i % N);
  Wt[(long)n * K + k] = f2bf(W[i]);
}

// continuous relative position bias table: (12, 49, 49) f32
__global__ void k_crpb(const float* __restrict__ w1, const float* __restrict__ b1,
                       const float* __restrict__ w2, const float* __restrict__ b2,
                       float* __restrict__ bias) {
  int p = blockIdx.x * 256 + threadIdx.x;
  if (p >= NTOK * NTOK) return;
  int i = p / NTOK, j = p % NTOK;
  float d0 = (float)(i / 7 - j / 7);
  float d1 = (float)(i % 7 - j % 7);
  d0 = copysignf(log1pf(fabsf(d0)), d0);
  d1 = copysignf(log1pf(fabsf(d1)), d1);
  float acc[NHEAD];
#pragma unroll
  for (int t = 0; t < NHEAD; ++t) acc[t] = b2[t];
  for (int hd = 0; hd < 64; ++hd) {
    float hv = fmaxf(d0 * w1[hd] + d1 * w1[64 + hd] + b1[hd], 0.0f);
#pragma unroll
    for (int t = 0; t < NHEAD; ++t) acc[t] += hv * w2[hd * NHEAD + t];
  }
#pragma unroll
  for (int t = 0; t < NHEAD; ++t) bias[(long)t * (NTOK * NTOK) + p] = acc[t];
}

// ---------------------------------------------------------------------------
// LN1 + cyclic shift(-3,-3) + window partition -> Xw bf16 (2048,64,384)
// one wave32 per token; grid 12544 x 256  (100352 waves)
// ---------------------------------------------------------------------------
__global__ __launch_bounds__(256)
void k_ln1_part(const float* __restrict__ x, const float* __restrict__ g,
                const float* __restrict__ b, u16* __restrict__ Xw) {
  int gt    = blockIdx.x * 256 + threadIdx.x;
  int token = gt >> 5;
  int lane  = gt & 31;
  int bidx = token / 3136, s = token % 3136;
  int h = s / 56, w = s % 56;
  const float* row = x + (long)token * DIMC;
  float v[12];
  float sum = 0.0f;
#pragma unroll
  for (int i = 0; i < 12; ++i) { v[i] = row[lane + i * 32]; sum += v[i]; }
#pragma unroll
  for (int off = 16; off > 0; off >>= 1) sum += __shfl_xor(sum, off, 32);
  float mean = sum * (1.0f / 384.0f);
  float vs = 0.0f;
#pragma unroll
  for (int i = 0; i < 12; ++i) { float d = v[i] - mean; vs += d * d; }
#pragma unroll
  for (int off = 16; off > 0; off >>= 1) vs += __shfl_xor(vs, off, 32);
  float rstd = rsqrtf(vs * (1.0f / 384.0f) + 1e-5f);
  // shifted coords: h' = (h-3) mod 56
  int hs = (h + 53) % 56, wsd = (w + 53) % 56;
  int wh = hs / 7, r = hs % 7, ww = wsd / 7, c = wsd % 7;
  long drow = ((long)bidx * 64 + wh * 8 + ww) * NPAD + (r * 7 + c);
  u16* dst = Xw + drow * DIMC;
#pragma unroll
  for (int i = 0; i < 12; ++i) {
    int col = lane + i * 32;
    dst[col] = f2bf((v[i] - mean) * rstd * g[col] + b[col]);
  }
}

// ---------------------------------------------------------------------------
// generic bf16 GEMM: C = A(MxK,bf16) @ Bt^T + bias ; Bt is (NxK) bf16
// block tile 128x128, 256 threads = 8 waves (4x2), wave tile 32x64
// double-buffered LDS filled with global_load_async_to_lds_b128 (ASYNCcnt)
// EPI: 0 = bias, store bf16
//      1 = bias, store f32
//      2 = bias + exact GELU, store bf16
//      3 = bias + residual(f32), store f32
// ---------------------------------------------------------------------------
template <int EPI>
__global__ __launch_bounds__(256)
void k_gemm_bf16(const u16* __restrict__ A, const u16* __restrict__ Bt,
                 const float* __restrict__ bias, void* __restrict__ Cout,
                 const float* __restrict__ resid, int K, int ldc) {
  __shared__ u16 As[2][128 * 40];
  __shared__ u16 Bs[2][128 * 40];
  const int tid  = threadIdx.x;
  const int lane = tid & 31;
  const int wv   = tid >> 5;
  const int waveM = wv >> 1;           // 0..3
  const int waveN = wv & 1;            // 0..1
  const long m0 = (long)blockIdx.x * 128;
  const long n0 = (long)blockIdx.y * 128;

  // per-thread tile-fill coordinates: 2 chunks each for A and B
  const int li0 = tid * 2, li1 = tid * 2 + 1;        // 0..511
  const int r0 = li0 >> 2, c0 = (li0 & 3) * 8;       // row, col-chunk (8 bf16)
  const int r1 = li1 >> 2, c1 = (li1 & 3) * 8;

  v8f acc[2][4];
#pragma unroll
  for (int i = 0; i < 2; ++i)
#pragma unroll
    for (int j = 0; j < 4; ++j)
#pragma unroll
      for (int r = 0; r < 8; ++r) acc[i][j][r] = 0.0f;

  const int nk = K >> 5;   // K/32 tiles

  // prologue: async-fill tile 0 into buffer 0
  {
    const u16* a0 = &A[(m0 + r0) * (long)K + c0];
    const u16* a1 = &A[(m0 + r1) * (long)K + c1];
    const u16* b0 = &Bt[(n0 + r0) * (long)K + c0];
    const u16* b1 = &Bt[(n0 + r1) * (long)K + c1];
    async_ld_b128(&As[0][r0 * 40 + c0], a0);
    async_ld_b128(&As[0][r1 * 40 + c1], a1);
    async_ld_b128(&Bs[0][r0 * 40 + c0], b0);
    async_ld_b128(&Bs[0][r1 * 40 + c1], b1);
  }

  for (int t = 0; t < nk; ++t) {
    const int cur = t & 1;
    if (t + 1 < nk) {
      // async-fill next tile while we (shortly) compute this one
      const int nxt = (t + 1) & 1;
      const int kt = (t + 1) << 5;
      async_ld_b128(&As[nxt][r0 * 40 + c0], &A[(m0 + r0) * (long)K + kt + c0]);
      async_ld_b128(&As[nxt][r1 * 40 + c1], &A[(m0 + r1) * (long)K + kt + c1]);
      async_ld_b128(&Bs[nxt][r0 * 40 + c0], &Bt[(n0 + r0) * (long)K + kt + c0]);
      async_ld_b128(&Bs[nxt][r1 * 40 + c1], &Bt[(n0 + r1) * (long)K + kt + c1]);
      wait_async4();   // current tile's 4 loads retired; next tile in flight
    } else {
      wait_async0();
    }
    __syncthreads();

    FragAB a[2];
#pragma unroll
    for (int ms = 0; ms < 2; ++ms) {
      int m  = waveM * 32 + ms * 16 + (lane & 15);
      int kb = (lane >> 4) * 16;
      a[ms].u[0] = *(const u32x4*)&As[cur][m * 40 + kb];
      a[ms].u[1] = *(const u32x4*)&As[cur][m * 40 + kb + 8];
    }
#pragma unroll
    for (int ns = 0; ns < 4; ++ns) {
      FragAB bf;
      int n    = waveN * 64 + ns * 16 + (lane & 15);
      int koff = (lane >> 4) * 16;
      bf.u[0] = *(const u32x4*)&Bs[cur][n * 40 + koff];
      bf.u[1] = *(const u32x4*)&Bs[cur][n * 40 + koff + 8];
#pragma unroll
      for (int ms = 0; ms < 2; ++ms)
        acc[ms][ns] = __builtin_amdgcn_wmma_f32_16x16x32_bf16(
            false, a[ms].v, false, bf.v, (short)0, acc[ms][ns], false, false);
    }
    __syncthreads();   // protect buffer reuse before next async fill lands
  }

#pragma unroll
  for (int ms = 0; ms < 2; ++ms)
#pragma unroll
    for (int ns = 0; ns < 4; ++ns) {
      long n = n0 + waveN * 64 + ns * 16 + (lane & 15);
      float bn = bias[n];
#pragma unroll
      for (int r = 0; r < 8; ++r) {
        long m = m0 + waveM * 32 + ms * 16 + ((lane >> 4) << 3) + r;
        float v = acc[ms][ns][r] + bn;
        if (EPI == 2) v = 0.5f * v * (1.0f + erff(v * 0.70710678118654752f));
        if (EPI == 3) v += resid[m * (long)ldc + n];
        if (EPI == 0 || EPI == 2)
          ((u16*)Cout)[m * (long)ldc + n] = f2bf(v);
        else
          ((float*)Cout)[m * (long)ldc + n] = v;
      }
    }
}

// ---------------------------------------------------------------------------
// cosine-attention prep: L2-normalize q,k per (row, head); fold tau into q
// ---------------------------------------------------------------------------
__global__ __launch_bounds__(256)
void k_qknorm(u16* __restrict__ qkv, const float* __restrict__ tau) {
  long idx = (long)blockIdx.x * 256 + threadIdx.x;
  if (idx >= (long)MROWS * 24) return;
  long row = idx / 24;
  int  t   = (int)(idx % 24);
  int part = t / NHEAD, head = t % NHEAD;
  u16* p = qkv + row * (3 * DIMC) + part * DIMC + head * HDIM;
  float vals[32];
  float ss = 0.0f;
#pragma unroll
  for (int q = 0; q < 4; ++q) {
    Pack8 pk;
    pk.u = *(const u32x4*)(p + q * 8);
#pragma unroll
    for (int e = 0; e < 8; ++e) {
      float f = bf2f(pk.s[e]);
      vals[q * 8 + e] = f;
      ss += f * f;
    }
  }
  float sc = 1.0f / fmaxf(sqrtf(ss), 1e-12f);
  if (part == 0) sc *= tau[head];
#pragma unroll
  for (int q = 0; q < 4; ++q) {
    Pack8 o;
#pragma unroll
    for (int e = 0; e < 8; ++e) o.s[e] = f2bf(vals[q * 8 + e] * sc);
    *(u32x4*)(p + q * 8) = o.u;
  }
}

// ---------------------------------------------------------------------------
// attention: one (window, head) per 128-thread block (4 waves)
// S = q k^T (raw scores from WMMA), bias+mask folded into the softmax pass,
// O = P V (WMMA). Region ids precomputed once into LDS.
// ---------------------------------------------------------------------------
__device__ __forceinline__ int swin_region(int wi, int p) {
  int gh = (wi >> 3) * 7 + p / 7;
  int gw = (wi & 7) * 7 + p % 7;
  int rh = gh < 49 ? 0 : (gh < 53 ? 1 : 2);
  int rw = gw < 49 ? 0 : (gw < 53 ? 1 : 2);
  return rh * 3 + rw;
}

__global__ __launch_bounds__(128)
void k_attn(const u16* __restrict__ qkv, const float* __restrict__ bias,
            u16* __restrict__ out) {
  __shared__ u16   Qs[64 * 40];
  __shared__ u16   Ks[64 * 40];
  __shared__ u16   Vt[32 * 72];   // transposed: Vt[d][token]
  __shared__ float S[64 * 65];
  __shared__ u16   P[64 * 72];
  __shared__ int   reg64[64];

  const int win  = blockIdx.x;
  const int head = blockIdx.y;
  const int wi   = win & 63;
  const int tid  = threadIdx.x;
  const int lane = tid & 31;
  const int wv   = tid >> 5;

  const u16* base = qkv + (long)win * NPAD * (3 * DIMC) + head * HDIM;
#pragma unroll
  for (int l = 0; l < 2; ++l) {
    int li = tid * 2 + l;               // 0..255
    int row = li >> 2, cb = li & 3;     // 64 rows x 4 chunks of 8
    long off = (long)row * (3 * DIMC) + cb * 8;
    *(u32x4*)&Qs[row * 40 + cb * 8] = *(const u32x4*)(base + off);
    *(u32x4*)&Ks[row * 40 + cb * 8] = *(const u32x4*)(base + off + DIMC);
    Pack8 pv;
    pv.u = *(const u32x4*)(base + off + 2 * DIMC);
#pragma unroll
    for (int e = 0; e < 8; ++e) Vt[(cb * 8 + e) * 72 + row] = pv.s[e];
  }
  if (tid < 64) reg64[tid] = (tid < NTOK) ? swin_region(wi, tid) : -1;
  __syncthreads();

  // ---- S = q k^T (raw scores) ----
  FragAB a;
  {
    int m  = wv * 16 + (lane & 15);
    int kb = (lane >> 4) * 16;
    a.u[0] = *(const u32x4*)&Qs[m * 40 + kb];
    a.u[1] = *(const u32x4*)&Qs[m * 40 + kb + 8];
  }
#pragma unroll
  for (int ns = 0; ns < 4; ++ns) {
    FragAB bf;
    int n    = ns * 16 + (lane & 15);
    int koff = (lane >> 4) * 16;
    bf.u[0] = *(const u32x4*)&Ks[n * 40 + koff];
    bf.u[1] = *(const u32x4*)&Ks[n * 40 + koff + 8];
    v8f accS;
#pragma unroll
    for (int r = 0; r < 8; ++r) accS[r] = 0.0f;
    accS = __builtin_amdgcn_wmma_f32_16x16x32_bf16(
        false, a.v, false, bf.v, (short)0, accS, false, false);
    int nn = ns * 16 + (lane & 15);
#pragma unroll
    for (int r = 0; r < 8; ++r) {
      int m = wv * 16 + ((lane >> 4) << 3) + r;
      S[m * 65 + nn] = accS[r];
    }
  }
  __syncthreads();

  // ---- bias + mask + softmax, one row per thread (rows 0..63) ----
  if (tid < 64) {
    const bool tok = tid < NTOK;
    const int  rt  = reg64[tid];
    const float* brow = bias + (long)head * (NTOK * NTOK) + (long)tid * NTOK;
    float mx = -3.4e38f;
    for (int j = 0; j < 64; ++j) {
      float s;
      if (tok && j < NTOK)
        s = S[tid * 65 + j] + brow[j] + ((rt == reg64[j]) ? 0.0f : -100.0f);
      else
        s = -1e30f;
      S[tid * 65 + j] = s;
      mx = fmaxf(mx, s);
    }
    float sm = 0.0f;
    for (int j = 0; j < 64; ++j) sm += expf(S[tid * 65 + j] - mx);
    float inv = 1.0f / sm;
    for (int j = 0; j < 64; ++j)
      P[tid * 72 + j] = f2bf(expf(S[tid * 65 + j] - mx) * inv);
  }
  __syncthreads();

  // ---- O = P @ V ----
  v8f acc2[2];
#pragma unroll
  for (int ns = 0; ns < 2; ++ns)
#pragma unroll
    for (int r = 0; r < 8; ++r) acc2[ns][r] = 0.0f;

#pragma unroll
  for (int kk = 0; kk < 2; ++kk) {
    FragAB pa;
    int m  = wv * 16 + (lane & 15);
    int kb = (lane >> 4) * 16;
    pa.u[0] = *(const u32x4*)&P[m * 72 + kk * 32 + kb];
    pa.u[1] = *(const u32x4*)&P[m * 72 + kk * 32 + kb + 8];
#pragma unroll
    for (int ns = 0; ns < 2; ++ns) {
      FragAB vb;
      int d    = ns * 16 + (lane & 15);
      int koff = kk * 32 + (lane >> 4) * 16;
      vb.u[0] = *(const u32x4*)&Vt[d * 72 + koff];
      vb.u[1] = *(const u32x4*)&Vt[d * 72 + koff + 8];
      acc2[ns] = __builtin_amdgcn_wmma_f32_16x16x32_bf16(
          false, pa.v, false, vb.v, (short)0, acc2[ns], false, false);
    }
  }
#pragma unroll
  for (int ns = 0; ns < 2; ++ns)
#pragma unroll
    for (int r = 0; r < 8; ++r) {
      int m = wv * 16 + ((lane >> 4) << 3) + r;
      if (m < NTOK) {
        int d = ns * 16 + (lane & 15);
        out[((long)win * NPAD + m) * DIMC + head * HDIM + d] = f2bf(acc2[ns][r]);
      }
    }
}

// ---------------------------------------------------------------------------
// window reverse + roll(+3,+3) + residual -> x1 (f32); LN2 -> bf16
// one wave per token
// ---------------------------------------------------------------------------
__global__ __launch_bounds__(256)
void k_resid_ln2(const float* __restrict__ x, const float* __restrict__ Y,
                 const float* __restrict__ g, const float* __restrict__ b,
                 float* __restrict__ x1, u16* __restrict__ ln2) {
  int gt    = blockIdx.x * 256 + threadIdx.x;
  int token = gt >> 5;
  int lane  = gt & 31;
  int bidx = token / 3136, s = token % 3136;
  int h = s / 56, w = s % 56;
  int hs = (h + 53) % 56, wsd = (w + 53) % 56;
  int wh = hs / 7, r = hs % 7, ww = wsd / 7, c = wsd % 7;
  long srow = ((long)bidx * 64 + wh * 8 + ww) * NPAD + (r * 7 + c);
  const float* xr = x + (long)token * DIMC;
  const float* yr = Y + srow * DIMC;
  float* xo = x1 + (long)token * DIMC;

  float v[12];
  float sum = 0.0f;
#pragma unroll
  for (int i = 0; i < 12; ++i) {
    int col = lane + i * 32;
    v[i] = xr[col] + yr[col];
    xo[col] = v[i];
    sum += v[i];
  }
#pragma unroll
  for (int off = 16; off > 0; off >>= 1) sum += __shfl_xor(sum, off, 32);
  float mean = sum * (1.0f / 384.0f);
  float vs = 0.0f;
#pragma unroll
  for (int i = 0; i < 12; ++i) { float d = v[i] - mean; vs += d * d; }
#pragma unroll
  for (int off = 16; off > 0; off >>= 1) vs += __shfl_xor(vs, off, 32);
  float rstd = rsqrtf(vs * (1.0f / 384.0f) + 1e-5f);
  u16* lo = ln2 + (long)token * DIMC;
#pragma unroll
  for (int i = 0; i < 12; ++i) {
    int col = lane + i * 32;
    lo[col] = f2bf((v[i] - mean) * rstd * g[col] + b[col]);
  }
}

// ---------------------------------------------------------------------------
// launch
// ---------------------------------------------------------------------------
extern "C" void kernel_launch(void* const* d_in, const int* in_sizes, int n_in,
                              void* d_out, int out_size, void* d_ws, size_t ws_size,
                              hipStream_t stream) {
  const float* x      = (const float*)d_in[0];
  const float* n1g    = (const float*)d_in[1];
  const float* n1b    = (const float*)d_in[2];
  const float* qkv_w  = (const float*)d_in[3];
  const float* qkv_b  = (const float*)d_in[4];
  const float* tau    = (const float*)d_in[5];
  const float* proj_w = (const float*)d_in[6];
  const float* proj_b = (const float*)d_in[7];
  const float* c_w1   = (const float*)d_in[8];
  const float* c_b1   = (const float*)d_in[9];
  const float* c_w2   = (const float*)d_in[10];
  const float* c_b2   = (const float*)d_in[11];
  const float* n2g    = (const float*)d_in[12];
  const float* n2b    = (const float*)d_in[13];
  const float* m_w1   = (const float*)d_in[14];
  const float* m_b1   = (const float*)d_in[15];
  const float* m_w2   = (const float*)d_in[16];
  const float* m_b2   = (const float*)d_in[17];

  char* ws = (char*)d_ws;
  u16*   wqkvT  = (u16*)(ws + OFF_WQKVT);
  u16*   wprojT = (u16*)(ws + OFF_WPROJT);
  u16*   w1T    = (u16*)(ws + OFF_W1T);
  u16*   w2T    = (u16*)(ws + OFF_W2T);
  float* crpb   = (float*)(ws + OFF_BIAS);
  u16*   Xw     = (u16*)(ws + OFF_XW);      // later aliased as attn_out
  u16*   qkvB   = (u16*)(ws + OFF_QKV);     // later aliased as Y (f32) / H (bf16)
  float* Yf     = (float*)(ws + OFF_QKV);
  u16*   Hb     = (u16*)(ws + OFF_QKV);
  float* x1     = (float*)(ws + OFF_X1);
  u16*   ln2b   = (u16*)(ws + OFF_LN2);

  // weight convert + transpose (f32 -> bf16, NxK)
  k_wtrans<<<(384 * 1152 + 255) / 256, 256, 0, stream>>>(qkv_w,  wqkvT, 384, 1152);
  k_wtrans<<<(384 * 384  + 255) / 256, 256, 0, stream>>>(proj_w, wprojT, 384, 384);
  k_wtrans<<<(384 * 1536 + 255) / 256, 256, 0, stream>>>(m_w1,   w1T, 384, 1536);
  k_wtrans<<<(1536 * 384 + 255) / 256, 256, 0, stream>>>(m_w2,   w2T, 1536, 384);

  // CRPB bias table
  k_crpb<<<(NTOK * NTOK + 255) / 256, 256, 0, stream>>>(c_w1, c_b1, c_w2, c_b2, crpb);

  // zero Xw (padding rows must be 0)
  k_zero16<<<2048, 256, 0, stream>>>((u32x4*)Xw, (long)TOTWIN * NPAD * DIMC * 2 / 16);

  // LN1 + shift + window partition
  k_ln1_part<<<TOKENS / 8, 256, 0, stream>>>(x, n1g, n1b, Xw);

  // QKV GEMM: (131072 x 384) @ (384 x 1152) -> bf16
  k_gemm_bf16<0><<<dim3(MROWS / 128, 1152 / 128), 256, 0, stream>>>(
      Xw, wqkvT, qkv_b, (void*)qkvB, nullptr, 384, 1152);

  // cosine-attention normalization (tau folded into q)
  {
    long tasks = (long)MROWS * 24;
    k_qknorm<<<(unsigned)((tasks + 255) / 256), 256, 0, stream>>>(qkvB, tau);
  }

  // attention per (window, head); writes attn_out over Xw buffer
  k_attn<<<dim3(TOTWIN, NHEAD), 128, 0, stream>>>(qkvB, crpb, Xw);

  // proj GEMM: (131072 x 384) @ (384 x 384) -> f32 Y (over qkv buffer)
  k_gemm_bf16<1><<<dim3(MROWS / 128, 384 / 128), 256, 0, stream>>>(
      Xw, wprojT, proj_b, (void*)Yf, nullptr, 384, 384);

  // window reverse + roll + residual -> x1 ; LN2 -> bf16
  k_resid_ln2<<<TOKENS / 8, 256, 0, stream>>>(x, Yf, n2g, n2b, x1, ln2b);

  // MLP GEMM1: (100352 x 384) @ (384 x 1536), GELU -> bf16 H (over qkv buffer)
  k_gemm_bf16<2><<<dim3(TOKENS / 128, 1536 / 128), 256, 0, stream>>>(
      ln2b, w1T, m_b1, (void*)Hb, nullptr, 384, 1536);

  // MLP GEMM2: (100352 x 1536) @ (1536 x 384) + x1 -> d_out f32
  k_gemm_bf16<3><<<dim3(TOKENS / 128, 384 / 128), 256, 0, stream>>>(
      Hb, w2T, m_b2, d_out, x1, 1536, 384);
}
